// LSTM_autoencoder_27315992003080
// MI455X (gfx1250) — matrix operand order repaired and checked by reading, compiled, not run on hardware
//
#include <hip/hip_runtime.h>
#include <stdint.h>

typedef __attribute__((ext_vector_type(16))) __bf16 v16bf;
typedef __attribute__((ext_vector_type(8)))  float  v8f;

#define RB 64      // persistent blocks per recurrence kernel
#define TPB 256    // threads per block (8 waves on wave32)

__device__ __forceinline__ uint16_t f2bf(float f) {
  union { float f; uint32_t u; } v; v.f = f;
  uint32_t r = v.u + 0x7FFFu + ((v.u >> 16) & 1u);   // round-to-nearest-even
  return (uint16_t)(r >> 16);
}

__device__ __forceinline__ float sigm(float x) { return 1.0f / (1.0f + __expf(-x)); }

__device__ __forceinline__ v16bf ldfrag(const uint16_t* p) {
  return *reinterpret_cast<const v16bf*>(p);
}

__device__ __forceinline__ v8f wmma_bf16(v16bf a, v16bf b, v8f c) {
  return __builtin_amdgcn_wmma_f32_16x16x32_bf16(false, a, false, b, (short)0, c, false, false);
}

// ---- grid-wide barrier (persistent kernel, monotonically increasing counter) ----
__device__ __forceinline__ void grid_barrier(int* cnt, int& phase, int nblocks) {
  __syncthreads();
  if (threadIdx.x == 0) {
    __threadfence();
    atomicAdd(cnt, 1);
    ++phase;
    const int target = phase * nblocks;
    while (atomicAdd(cnt, 0) < target) { __builtin_amdgcn_s_sleep(1); }
    __threadfence();
  }
  __syncthreads();
}

// ---- scratch zeroing ----
__global__ void zero_kernel(uint32_t* p, int n) {
  for (int i = blockIdx.x * blockDim.x + threadIdx.x; i < n; i += gridDim.x * blockDim.x)
    p[i] = 0u;
}

// ---- pack W [N][K] row-major fp32 -> bf16 WMMA B-fragment tiles (tile = (kt*NT+nt), 512 elems) ----
__global__ void pack_b_kernel(const float* __restrict__ W, uint16_t* __restrict__ dst,
                              int N, int K) {
  const int NT = N >> 4;
  const int total = N * K;
  for (int idx = blockIdx.x * blockDim.x + threadIdx.x; idx < total;
       idx += gridDim.x * blockDim.x) {
    int e    = idx & 15;
    int lane = (idx >> 4) & 31;
    int tile = idx >> 9;
    int nt   = tile % NT;
    int kt   = tile / NT;
    int n = nt * 16 + (lane & 15);
    int k = kt * 32 + ((e < 8) ? e : (e + 8)) + ((lane >= 16) ? 8 : 0);
    dst[idx] = f2bf(W[n * K + k]);
  }
}

// ---- pack x [B=128][T=128][64] -> per-t A-fragment tiles (KT=2) ----
__global__ void pack_x_kernel(const float* __restrict__ x, uint16_t* __restrict__ dst) {
  const int total = 128 * 128 * 64;
  for (int idx = blockIdx.x * blockDim.x + threadIdx.x; idx < total;
       idx += gridDim.x * blockDim.x) {
    int e    = idx & 15;
    int lane = (idx >> 4) & 31;
    int tile = (idx >> 9) & 15;   // 16 tiles per timestep
    int t    = idx >> 13;
    int mt = tile >> 1;
    int kt = tile & 1;
    int b = mt * 16 + (lane & 15);
    int k = (lane < 16) ? ((e < 8) ? e : (e + 8)) : ((e < 8) ? (e + 8) : (e + 16));
    k += kt * 32;
    dst[idx] = f2bf(x[(b * 128 + t) * 64 + k]);
  }
}

// ---- persistent LSTM recurrence: g = h@Whh^T (+ xseq[t]@Wih^T) (+ gconst) + biases ----
// Block -> fixed N-tile pair (weights cached in LDS once); wave-in-block -> M-tile.
__global__ void __launch_bounds__(TPB) lstm_recur_kernel(
    const uint16_t* __restrict__ Whh_p,   // B-packed [K=512, N=2048]
    const uint16_t* __restrict__ Wih_p,   // B-packed [K=Kin, N=2048] or null
    const uint16_t* __restrict__ xseq_p,  // per-t A-packed input, stride 128*Kin, or null
    const float*    __restrict__ gconst,  // [128*2048] or null (decoder L0 constant input)
    const float*    __restrict__ bih,
    const float*    __restrict__ bhh,
    uint16_t*       __restrict__ hcur_p,  // A-packed h [128*512], zero-initialized
    float*          __restrict__ cstate,  // [128*512], zero-initialized
    uint16_t*       __restrict__ hseq_p,  // per-t A-packed h out, stride 65536, or null
    float*          __restrict__ gbuf,    // [128*2048] scratch
    int* sync_cnt, int Kin, int T) {
  __shared__ uint16_t sWhh[16384];   // 2 N-tiles x 16 K-tiles  (32 KB)
  __shared__ uint16_t sWih[16384];   // 2 N-tiles x <=16 K-tiles (32 KB)

  const int lane = threadIdx.x & 31;
  const int m    = threadIdx.x >> 5;        // M-tile 0..7 (wave in block)
  const int nt0  = blockIdx.x * 2;          // N-tiles handled by this block
  const int KT2  = Kin >> 5;
  int phase = 0;

  // ---- one-time weight preload into LDS (tiles nt0, nt0+1 are contiguous) ----
  {
    const uint4* src = (const uint4*)Whh_p;  // 8 uint16 per uint4
    uint4* dst = (uint4*)sWhh;
    for (int i = threadIdx.x; i < 16 * 128; i += TPB) {
      int kt = i >> 7, r = i & 127;
      dst[i] = src[(size_t)(kt * 128 + nt0) * 64 + r];
    }
    if (Wih_p) {
      const uint4* src2 = (const uint4*)Wih_p;
      uint4* dst2 = (uint4*)sWih;
      for (int i = threadIdx.x; i < KT2 * 128; i += TPB) {
        int kt = i >> 7, r = i & 127;
        dst2[i] = src2[(size_t)(kt * 128 + nt0) * 64 + r];
      }
    }
    __syncthreads();
  }

  for (int t = 0; t < T; ++t) {
    v8f acc0 = {}; v8f acc1 = {};
    // recurrent GEMM: h[128x512] @ Whh^T ; B fragments from LDS
    const uint16_t* ap = hcur_p + (m * 16) * 512 + lane * 16;
    #pragma unroll 4
    for (int kt = 0; kt < 16; ++kt) {
      v16bf a  = ldfrag(ap + kt * 512);
      v16bf b0 = ldfrag(sWhh + kt * 1024 + lane * 16);
      v16bf b1 = ldfrag(sWhh + kt * 1024 + 512 + lane * 16);
      acc0 = wmma_bf16(a, b0, acc0);
      acc1 = wmma_bf16(a, b1, acc1);
    }
    // fused input GEMM from previous layer's stored sequence
    if (Wih_p) {
      const uint16_t* xp = xseq_p + (size_t)t * (128 * Kin) + (m * KT2) * 512 + lane * 16;
      for (int kt = 0; kt < KT2; ++kt) {
        v16bf a  = ldfrag(xp + kt * 512);
        v16bf b0 = ldfrag(sWih + kt * 1024 + lane * 16);
        v16bf b1 = ldfrag(sWih + kt * 1024 + 512 + lane * 16);
        acc0 = wmma_bf16(a, b0, acc0);
        acc1 = wmma_bf16(a, b1, acc1);
      }
    }
    // store g tiles (C-fragment layout -> row-major gbuf)
    {
      const int col0  = nt0 * 16 + (lane & 15);
      const int rbase = m * 16 + ((lane >= 16) ? 8 : 0);
      #pragma unroll
      for (int v = 0; v < 8; ++v) {
        gbuf[(rbase + v) * 2048 + col0]      = acc0[v];
        gbuf[(rbase + v) * 2048 + col0 + 16] = acc1[v];
      }
    }
    grid_barrier(sync_cnt, phase, RB);

    // elementwise gates: 128*512 items over 16384 threads
    for (int idx = blockIdx.x * TPB + threadIdx.x; idx < 128 * 512; idx += RB * TPB) {
      int b = idx >> 9;
      int j = idx & 511;
      const float* gr = gbuf + b * 2048;
      float gi = gr[j]        + bih[j]        + bhh[j];
      float gf = gr[512 + j]  + bih[512 + j]  + bhh[512 + j];
      float gg = gr[1024 + j] + bih[1024 + j] + bhh[1024 + j];
      float go = gr[1536 + j] + bih[1536 + j] + bhh[1536 + j];
      if (gconst) {
        const float* gc = gconst + b * 2048;
        gi += gc[j]; gf += gc[512 + j]; gg += gc[1024 + j]; go += gc[1536 + j];
      }
      float c = cstate[idx];
      c = sigm(gf) * c + sigm(gi) * tanhf(gg);
      cstate[idx] = c;
      float h = sigm(go) * tanhf(c);
      uint16_t hb = f2bf(h);
      // write h into A-fragment layout for next step / next layer
      int mt = b >> 4, r = b & 15;
      int kt = j >> 5, kk = j & 31;
      int ln, e;
      if      (kk < 8)  { ln = r;      e = kk;      }
      else if (kk < 16) { ln = r + 16; e = kk - 8;  }
      else if (kk < 24) { ln = r;      e = kk - 8;  }
      else              { ln = r + 16; e = kk - 16; }
      int off = ((mt * 16 + kt) << 9) + (ln << 4) + e;
      hcur_p[off] = hb;
      if (hseq_p) hseq_p[(size_t)t * 65536 + off] = hb;
    }
    grid_barrier(sync_cnt, phase, RB);
  }
}

// ---- one-shot GEMM: out[128 x NT*16] = A(packed) @ B(packed), fp32 out ----
__global__ void __launch_bounds__(TPB) gemm_ab_kernel(
    const uint16_t* __restrict__ Ap, const uint16_t* __restrict__ Bp,
    float* __restrict__ out, int KT, int NT) {
  const int lane = threadIdx.x & 31;
  const int wave = blockIdx.x * (blockDim.x >> 5) + (threadIdx.x >> 5);
  const int m  = wave / NT;
  const int nt = wave % NT;
  v8f acc = {};
  const uint16_t* ap = Ap + (m * KT) * 512 + lane * 16;
  for (int kt = 0; kt < KT; ++kt) {
    v16bf a = ldfrag(ap + kt * 512);
    v16bf b = ldfrag(Bp + (size_t)(kt * NT + nt) * 512 + lane * 16);
    acc = wmma_bf16(a, b, acc);
  }
  const int col   = nt * 16 + (lane & 15);
  const int rbase = m * 16 + ((lane >= 16) ? 8 : 0);
  const int N = NT * 16;
  #pragma unroll
  for (int v = 0; v < 8; ++v) out[(rbase + v) * N + col] = acc[v];
}

// ---- final projection: out[b][t][i] = hs[t][b][:] . fcW[i][:] + fcb[i] ----
__global__ void __launch_bounds__(TPB) fc_kernel(
    const uint16_t* __restrict__ hseq_p, const uint16_t* __restrict__ fcWp,
    const float* __restrict__ fcb, float* __restrict__ out) {
  const int lane = threadIdx.x & 31;
  const int wave = blockIdx.x * 8 + (threadIdx.x >> 5);  // 0..4095
  const int t  = wave >> 5;
  const int r  = wave & 31;
  const int m  = r >> 2;
  const int nt = r & 3;
  v8f acc = {};
  const uint16_t* ap = hseq_p + (size_t)t * 65536 + (m * 16) * 512 + lane * 16;
  #pragma unroll 4
  for (int kt = 0; kt < 16; ++kt) {
    v16bf a = ldfrag(ap + kt * 512);
    v16bf b = ldfrag(fcWp + (size_t)(kt * 4 + nt) * 512 + lane * 16);
    acc = wmma_bf16(a, b, acc);
  }
  const int n    = nt * 16 + (lane & 15);
  const float bv = fcb[n];
  const int bb   = m * 16 + ((lane >= 16) ? 8 : 0);
  #pragma unroll
  for (int v = 0; v < 8; ++v)
    out[((size_t)(bb + v) * 128 + t) * 64 + n] = acc[v] + bv;
}

extern "C" void kernel_launch(void* const* d_in, const int* in_sizes, int n_in,
                              void* d_out, int out_size, void* d_ws, size_t ws_size,
                              hipStream_t stream) {
  (void)in_sizes; (void)n_in; (void)out_size; (void)ws_size;
  const float* x     = (const float*)d_in[0];
  const float* eWih0 = (const float*)d_in[1];
  const float* eWhh0 = (const float*)d_in[2];
  const float* ebih0 = (const float*)d_in[3];
  const float* ebhh0 = (const float*)d_in[4];
  const float* eWih1 = (const float*)d_in[5];
  const float* eWhh1 = (const float*)d_in[6];
  const float* ebih1 = (const float*)d_in[7];
  const float* ebhh1 = (const float*)d_in[8];
  const float* dWih0 = (const float*)d_in[9];
  const float* dWhh0 = (const float*)d_in[10];
  const float* dbih0 = (const float*)d_in[11];
  const float* dbhh0 = (const float*)d_in[12];
  const float* dWih1 = (const float*)d_in[13];
  const float* dWhh1 = (const float*)d_in[14];
  const float* dbih1 = (const float*)d_in[15];
  const float* dbhh1 = (const float*)d_in[16];
  const float* fcW   = (const float*)d_in[17];
  const float* fcb   = (const float*)d_in[18];
  float* out = (float*)d_out;

  size_t off = 0;
  auto alloc = [&](size_t bytes) -> void* {
    size_t o = (off + 255) & ~(size_t)255;
    off = o + bytes;
    return (void*)((char*)d_ws + o);
  };
  // zero-initialized state region (contiguous from offset 0)
  int*      syncc = (int*)alloc(1024);          // 4 counters, 256B apart
  uint16_t* h_e0  = (uint16_t*)alloc(131072);
  uint16_t* h_e1  = (uint16_t*)alloc(131072);
  uint16_t* h_d0  = (uint16_t*)alloc(131072);
  uint16_t* h_d1  = (uint16_t*)alloc(131072);
  float*    c_e0  = (float*)alloc(262144);
  float*    c_e1  = (float*)alloc(262144);
  float*    c_d0  = (float*)alloc(262144);
  float*    c_d1  = (float*)alloc(262144);
  const size_t zero_bytes = off;
  // non-zeroed scratch
  float*    gbuf  = (float*)alloc((size_t)128 * 2048 * 4);
  float*    gz0   = (float*)alloc((size_t)128 * 2048 * 4);
  uint16_t* xp    = (uint16_t*)alloc((size_t)128 * 128 * 64 * 2);
  uint16_t* y0seq = (uint16_t*)alloc((size_t)128 * 65536 * 2);
  uint16_t* h0seq = (uint16_t*)alloc((size_t)128 * 65536 * 2);
  uint16_t* h1seq = (uint16_t*)alloc((size_t)128 * 65536 * 2);
  uint16_t* eWih0p = (uint16_t*)alloc((size_t)2048 * 64 * 2);
  uint16_t* eWhh0p = (uint16_t*)alloc((size_t)2048 * 512 * 2);
  uint16_t* eWih1p = (uint16_t*)alloc((size_t)2048 * 512 * 2);
  uint16_t* eWhh1p = (uint16_t*)alloc((size_t)2048 * 512 * 2);
  uint16_t* dWih0p = (uint16_t*)alloc((size_t)2048 * 512 * 2);
  uint16_t* dWhh0p = (uint16_t*)alloc((size_t)2048 * 512 * 2);
  uint16_t* dWih1p = (uint16_t*)alloc((size_t)2048 * 512 * 2);
  uint16_t* dWhh1p = (uint16_t*)alloc((size_t)2048 * 512 * 2);
  uint16_t* fcWp   = (uint16_t*)alloc((size_t)64 * 512 * 2);

  zero_kernel<<<64, TPB, 0, stream>>>((uint32_t*)d_ws, (int)(zero_bytes / 4));

  pack_b_kernel<<<1024, TPB, 0, stream>>>(eWih0, eWih0p, 2048, 64);
  pack_b_kernel<<<1024, TPB, 0, stream>>>(eWhh0, eWhh0p, 2048, 512);
  pack_b_kernel<<<1024, TPB, 0, stream>>>(eWih1, eWih1p, 2048, 512);
  pack_b_kernel<<<1024, TPB, 0, stream>>>(eWhh1, eWhh1p, 2048, 512);
  pack_b_kernel<<<1024, TPB, 0, stream>>>(dWih0, dWih0p, 2048, 512);
  pack_b_kernel<<<1024, TPB, 0, stream>>>(dWhh0, dWhh0p, 2048, 512);
  pack_b_kernel<<<1024, TPB, 0, stream>>>(dWih1, dWih1p, 2048, 512);
  pack_b_kernel<<<1024, TPB, 0, stream>>>(dWhh1, dWhh1p, 2048, 512);
  pack_b_kernel<<<256,  TPB, 0, stream>>>(fcW, fcWp, 64, 512);
  pack_x_kernel<<<1024, TPB, 0, stream>>>(x, xp);

  // encoder layer 0: input = x (Kin=64), store full output sequence
  lstm_recur_kernel<<<RB, TPB, 0, stream>>>(eWhh0p, eWih0p, xp, nullptr, ebih0, ebhh0,
                                            h_e0, c_e0, y0seq, gbuf, syncc + 0, 64, 128);
  // encoder layer 1: input = y0seq (Kin=512); only final h needed (= z, left in h_e1)
  lstm_recur_kernel<<<RB, TPB, 0, stream>>>(eWhh1p, eWih1p, y0seq, nullptr, ebih1, ebhh1,
                                            h_e1, c_e1, nullptr, gbuf, syncc + 64, 512, 128);
  // decoder constant-input GEMM: gz0 = z @ dWih0^T
  gemm_ab_kernel<<<128, TPB, 0, stream>>>(h_e1, dWih0p, gz0, 16, 128);
  // decoder layer 0: constant input (gconst), store h0 sequence
  lstm_recur_kernel<<<RB, TPB, 0, stream>>>(dWhh0p, nullptr, nullptr, gz0, dbih0, dbhh0,
                                            h_d0, c_d0, h0seq, gbuf, syncc + 128, 0, 128);
  // decoder layer 1: input = h0seq (Kin=512), store h1 sequence
  lstm_recur_kernel<<<RB, TPB, 0, stream>>>(dWhh1p, dWih1p, h0seq, nullptr, dbih1, dbhh1,
                                            h_d1, c_d1, h1seq, gbuf, syncc + 192, 512, 128);
  // output projection
  fc_kernel<<<512, TPB, 0, stream>>>(h1seq, fcWp, fcb, out);
}
